// OSPNextSelfAttention_23880018166204
// MI455X (gfx1250) — matrix operand order repaired and checked by reading, not compiled, not run on hardware
//
#include <hip/hip_runtime.h>
#include <hip/hip_bf16.h>

typedef __attribute__((ext_vector_type(16))) __bf16 v16bf;
typedef __attribute__((ext_vector_type(8)))  __bf16 v8bf;
typedef __attribute__((ext_vector_type(8)))  float  v8f;
typedef __attribute__((ext_vector_type(4)))  int    v4i;

union Frag16 { v16bf v; v8bf h[2]; };

#define WMMA_BF16(A, B, C) \
    __builtin_amdgcn_wmma_f32_16x16x32_bf16(false, (A), false, (B), (short)0, (C), false, false)

// ---------------------------------------------------------------------------
// Async global -> LDS copy (16 bytes per lane).  gfx1250 GLOBAL_LOAD_ASYNC_
// TO_LDS_B128 is tracked with ASYNCcnt; fall back to plain load/store if the
// builtin is unavailable in this toolchain.  The builtin's parameters are
// int4 pointers in the global (AS1) / LDS (AS3) address spaces.
// ---------------------------------------------------------------------------
#if __has_builtin(__builtin_amdgcn_global_load_async_to_lds_b128)
#define HAVE_ASYNC_LDS 1
__device__ __forceinline__ void async_copy16(const __bf16* g, __bf16* l) {
    __builtin_amdgcn_global_load_async_to_lds_b128(
        (__attribute__((address_space(1))) v4i*)(g),
        (__attribute__((address_space(3))) v4i*)(l), 0, 0);
}
#else
#define HAVE_ASYNC_LDS 0
__device__ __forceinline__ void async_copy16(const __bf16* g, __bf16* l) {
    *(v8bf*)l = *(const v8bf*)g;
}
#endif

__device__ __forceinline__ void async_wait0() {
#if HAVE_ASYNC_LDS
#if __has_builtin(__builtin_amdgcn_s_wait_asynccnt)
    __builtin_amdgcn_s_wait_asynccnt(0);
#else
    asm volatile("s_wait_asynccnt 0" ::: "memory");
#endif
#endif
}

// ---------------------------------------------------------------------------
// f32 -> bf16 elementwise convert (vectorized, 8 elems/thread)
// ---------------------------------------------------------------------------
__global__ void convert_f32_to_bf16(const float* __restrict__ in,
                                    __bf16* __restrict__ out, int n8) {
    int i = blockIdx.x * blockDim.x + threadIdx.x;
    if (i >= n8) return;
    const float4* p = (const float4*)in + (size_t)i * 2;
    float4 a = p[0], b = p[1];
    v8bf o;
    o[0] = (__bf16)a.x; o[1] = (__bf16)a.y; o[2] = (__bf16)a.z; o[3] = (__bf16)a.w;
    o[4] = (__bf16)b.x; o[5] = (__bf16)b.y; o[6] = (__bf16)b.z; o[7] = (__bf16)b.w;
    *((v8bf*)out + i) = o;
}

// ---------------------------------------------------------------------------
// w (R x C, f32, row-major) -> wT (C x R, bf16, row-major)
// ---------------------------------------------------------------------------
__global__ void transpose_convert(const float* __restrict__ in,
                                  __bf16* __restrict__ out, int R, int C) {
    __shared__ __align__(16) __bf16 tile[32][33];
    int bx = blockIdx.x * 32;
    int by = blockIdx.y * 32;
    for (int i = threadIdx.y; i < 32; i += 8)
        tile[i][threadIdx.x] = (__bf16)in[(size_t)(by + i) * C + bx + threadIdx.x];
    __syncthreads();
    for (int i = threadIdx.y; i < 32; i += 8)
        out[(size_t)(bx + i) * R + by + threadIdx.x] = tile[threadIdx.x][i];
}

// ---------------------------------------------------------------------------
// C[M,N] = A[M,K](bf16) @ BT[N,K](bf16)^T + bias[N]
// 128x128 block tile, 4 waves (128 thr) in 2x2, each wave 64x64 = 4x4 WMMA
// tiles.  LDS is double-buffered; tiles staged with async-to-LDS copies that
// overlap the WMMA stream (one barrier per K-step).
// ---------------------------------------------------------------------------
__device__ __forceinline__ void stage_tiles(const __bf16* __restrict__ A,
                                            const __bf16* __restrict__ BT,
                                            __bf16 (*As)[40], __bf16 (*Bs)[40],
                                            int m0, int n0, int K, int k0, int tid) {
#pragma unroll
    for (int i = 0; i < 4; ++i) {
        int e   = tid + i * 128;      // 0..511
        int row = e >> 2;             // 0..127
        int cg  = (e & 3) << 3;       // 0,8,16,24
        async_copy16(&A [(size_t)(m0 + row) * K + k0 + cg], &As[row][cg]);
        async_copy16(&BT[(size_t)(n0 + row) * K + k0 + cg], &Bs[row][cg]);
    }
}

template <bool OUT_BF16>
__global__ __launch_bounds__(128)
void gemm_bf16_wmma(const __bf16* __restrict__ A, const __bf16* __restrict__ BT,
                    const float* __restrict__ bias, void* __restrict__ Cout,
                    int M, int N, int K) {
    __shared__ __align__(16) __bf16 As[2][128][40];   // 32 K + pad, double buffered
    __shared__ __align__(16) __bf16 Bs[2][128][40];

    const int m0   = blockIdx.y * 128;
    const int n0   = blockIdx.x * 128;
    const int tid  = threadIdx.x;
    const int lane = tid & 31;
    const int wave = tid >> 5;
    const int wm   = (wave >> 1) * 64;
    const int wn   = (wave & 1) * 64;
    const int l15  = lane & 15;
    const int half = lane >> 4;

    v8f acc[4][4];
#pragma unroll
    for (int i = 0; i < 4; ++i)
#pragma unroll
        for (int j = 0; j < 4; ++j)
            acc[i][j] = (v8f){0.f, 0.f, 0.f, 0.f, 0.f, 0.f, 0.f, 0.f};

    const int ksteps = K >> 5;
    stage_tiles(A, BT, As[0], Bs[0], m0, n0, K, 0, tid);

    for (int ks = 0; ks < ksteps; ++ks) {
        const int cur = ks & 1;
        async_wait0();
        __syncthreads();
        if (ks + 1 < ksteps)
            stage_tiles(A, BT, As[1 - cur], Bs[1 - cur], m0, n0, K, (ks + 1) << 5, tid);

        Frag16 af[4];
#pragma unroll
        for (int i = 0; i < 4; ++i) {
            int r = wm + i * 16 + l15;
            af[i].h[0] = *(const v8bf*)&As[cur][r][half * 8];
            af[i].h[1] = *(const v8bf*)&As[cur][r][16 + half * 8];
        }
#pragma unroll
        for (int j = 0; j < 4; ++j) {
            Frag16 bfr;
            int c = wn + j * 16 + l15;
            bfr.h[0] = *(const v8bf*)&Bs[cur][c][half * 8];
            bfr.h[1] = *(const v8bf*)&Bs[cur][c][16 + half * 8];
#pragma unroll
            for (int i = 0; i < 4; ++i)
                acc[i][j] = WMMA_BF16(af[i].v, bfr.v, acc[i][j]);
        }
    }

    // epilogue: C layout lane=col(n), vgpr r = row (r + half*8)
#pragma unroll
    for (int j = 0; j < 4; ++j) {
        int n = n0 + wn + j * 16 + l15;
        float bv = bias[n];
#pragma unroll
        for (int i = 0; i < 4; ++i) {
#pragma unroll
            for (int r = 0; r < 8; ++r) {
                int m = m0 + wm + i * 16 + r + half * 8;
                float val = acc[i][j][r] + bv;
                if (OUT_BF16)
                    ((__bf16*)Cout)[(size_t)m * N + n] = (__bf16)val;
                else
                    ((float*)Cout)[(size_t)m * N + n] = val;
            }
        }
    }
}

// ---------------------------------------------------------------------------
// Flash attention over skiparse subsequences.
// qkv: bf16 [4096, 3072] rows = tokens, cols = [Q|K|V] each 1024 (16 heads x 64)
// token for (j,g) = j + 4*g.  One block per (j, h, 64-query tile); 4 waves,
// each wave owns 16 queries; stream keys in tiles of 32 with online softmax.
// ---------------------------------------------------------------------------
__global__ __launch_bounds__(128)
void flash_attn(const __bf16* __restrict__ qkv, __bf16* __restrict__ attn) {
    constexpr int C = 1024, C3 = 3072, D = 64, S = 4;
    __shared__ __align__(16) __bf16 Kt[32][72];       // [key][d] + pad
    __shared__ __align__(16) __bf16 Vt[64][40];       // [d][key] (transposed) + pad
    __shared__ __align__(16) __bf16 Pb[4][16][40];    // per-wave P staging

    const int bid  = blockIdx.x;        // ((j*16)+h)*16 + qt
    const int qt   = bid & 15;
    const int h    = (bid >> 4) & 15;
    const int j    = bid >> 8;
    const int tid  = threadIdx.x;
    const int lane = tid & 31;
    const int wave = tid >> 5;
    const int l15  = lane & 15;
    const int half = lane >> 4;
    const int qbase = qt * 64 + wave * 16;

    // Q fragments (held in registers for the whole key loop): rows g=qbase+l15
    Frag16 qf[2];
    {
        int g = qbase + l15;
        size_t base = (size_t)(j + S * g) * C3 + h * D;
#pragma unroll
        for (int f = 0; f < 2; ++f) {
            qf[f].h[0] = *(const v8bf*)&qkv[base + f * 32 + half * 8];
            qf[f].h[1] = *(const v8bf*)&qkv[base + f * 32 + 16 + half * 8];
        }
    }

    v8f O[4];
#pragma unroll
    for (int t = 0; t < 4; ++t) O[t] = (v8f){0.f,0.f,0.f,0.f,0.f,0.f,0.f,0.f};
    float mrow[8], lrow[8];
#pragma unroll
    for (int r = 0; r < 8; ++r) { mrow[r] = -1e30f; lrow[r] = 0.f; }

    const float scale = 0.125f;   // 1/sqrt(64)

    for (int kt = 0; kt < 32; ++kt) {
        // cooperative load: K tile async straight into LDS, V tile transposed
#pragma unroll
        for (int i = 0; i < 2; ++i) {
            int e   = tid + i * 128;   // 0..255
            int key = e >> 3;          // 0..31
            int dg  = (e & 7) << 3;    // 0..56
            int g   = kt * 32 + key;
            size_t base = (size_t)(j + S * g) * C3 + h * D + dg;
            async_copy16(&qkv[base + C], &Kt[key][dg]);                 // K section
            v8bf vv = *(const v8bf*)&qkv[base + 2 * C];                 // V section
#pragma unroll
            for (int q2 = 0; q2 < 8; ++q2) Vt[dg + q2][key] = vv[q2];
        }
        async_wait0();
        __syncthreads();

        // S = Q @ K^T : two 16x16 score tiles (keys 0-15 and 16-31), K-dim = d = 2x32
        v8f s0 = (v8f){0.f,0.f,0.f,0.f,0.f,0.f,0.f,0.f};
        v8f s1 = s0;
#pragma unroll
        for (int f = 0; f < 2; ++f) {
            Frag16 kb0, kb1;
            kb0.h[0] = *(const v8bf*)&Kt[l15][f * 32 + half * 8];
            kb0.h[1] = *(const v8bf*)&Kt[l15][f * 32 + 16 + half * 8];
            kb1.h[0] = *(const v8bf*)&Kt[16 + l15][f * 32 + half * 8];
            kb1.h[1] = *(const v8bf*)&Kt[16 + l15][f * 32 + 16 + half * 8];
            s0 = WMMA_BF16(qf[f].v, kb0.v, s0);
            s1 = WMMA_BF16(qf[f].v, kb1.v, s1);
        }

        // online softmax: row (r + half*8) spans 16 lanes of this half -> xor reduce
        float corrv[8];
#pragma unroll
        for (int r = 0; r < 8; ++r) {
            float a = s0[r] * scale, b = s1[r] * scale;
            float mx = fmaxf(a, b);
            mx = fmaxf(mx, __shfl_xor(mx, 1, 32));
            mx = fmaxf(mx, __shfl_xor(mx, 2, 32));
            mx = fmaxf(mx, __shfl_xor(mx, 4, 32));
            mx = fmaxf(mx, __shfl_xor(mx, 8, 32));
            float mnew = fmaxf(mrow[r], mx);
            float corr = __expf(mrow[r] - mnew);
            mrow[r] = mnew;
            float p0 = __expf(a - mnew);
            float p1 = __expf(b - mnew);
            s0[r] = p0; s1[r] = p1;
            float rs = p0 + p1;
            rs += __shfl_xor(rs, 1, 32);
            rs += __shfl_xor(rs, 2, 32);
            rs += __shfl_xor(rs, 4, 32);
            rs += __shfl_xor(rs, 8, 32);
            lrow[r] = lrow[r] * corr + rs;
            corrv[r] = corr;
        }
#pragma unroll
        for (int t = 0; t < 4; ++t)
#pragma unroll
            for (int r = 0; r < 8; ++r) O[t][r] *= corrv[r];

        // P (C layout) -> LDS -> A-fragment layout (wave-private, no barrier needed)
#pragma unroll
        for (int r = 0; r < 8; ++r) {
            Pb[wave][r + half * 8][l15]      = (__bf16)s0[r];
            Pb[wave][r + half * 8][16 + l15] = (__bf16)s1[r];
        }
        Frag16 pf;
        pf.h[0] = *(const v8bf*)&Pb[wave][l15][half * 8];
        pf.h[1] = *(const v8bf*)&Pb[wave][l15][16 + half * 8];

        // O += P @ V : 4 output 16x16 tiles over d, K-dim = 32 keys
#pragma unroll
        for (int t = 0; t < 4; ++t) {
            Frag16 vb;
            vb.h[0] = *(const v8bf*)&Vt[t * 16 + l15][half * 8];
            vb.h[1] = *(const v8bf*)&Vt[t * 16 + l15][16 + half * 8];
            O[t] = WMMA_BF16(pf.v, vb.v, O[t]);
        }
        __syncthreads();
    }

    // normalize and scatter back to token order (bf16 for the output GEMM)
    float inv[8];
#pragma unroll
    for (int r = 0; r < 8; ++r) inv[r] = 1.f / lrow[r];
#pragma unroll
    for (int t = 0; t < 4; ++t) {
#pragma unroll
        for (int r = 0; r < 8; ++r) {
            int g = qbase + r + half * 8;
            int col = h * D + t * 16 + l15;
            attn[(size_t)(j + S * g) * C + col] = (__bf16)(O[t][r] * inv[r]);
        }
    }
}

// ---------------------------------------------------------------------------
// launch
// ---------------------------------------------------------------------------
extern "C" void kernel_launch(void* const* d_in, const int* in_sizes, int n_in,
                              void* d_out, int out_size, void* d_ws, size_t ws_size,
                              hipStream_t stream) {
    const float* x     = (const float*)d_in[0];   // [4096,1024]
    const float* w_qkv = (const float*)d_in[1];   // [1024,3072]
    const float* b_qkv = (const float*)d_in[2];   // [3072]
    const float* w_o   = (const float*)d_in[3];   // [1024,1024]
    const float* b_o   = (const float*)d_in[4];   // [1024]
    float* out = (float*)d_out;                   // [4096,1024]

    const int M = 4096, C = 1024, C3 = 3072;
    char* ws = (char*)d_ws;
    __bf16* x_bf    = (__bf16*)(ws);                              //  8 MB
    __bf16* wqkvT   = (__bf16*)(ws + (size_t)8  * 1024 * 1024);   //  6 MB
    __bf16* w_oT    = (__bf16*)(ws + (size_t)14 * 1024 * 1024);   //  2 MB
    __bf16* qkv_bf  = (__bf16*)(ws + (size_t)16 * 1024 * 1024);   // 24 MB
    __bf16* attn_bf = (__bf16*)(ws + (size_t)40 * 1024 * 1024);   //  8 MB

    // 1) convert activations to bf16
    {
        int n8 = M * C / 8;
        convert_f32_to_bf16<<<dim3((n8 + 255) / 256), dim3(256), 0, stream>>>(x, x_bf, n8);
    }
    // 2) transpose+convert weights (so GEMM B operand is K-major)
    transpose_convert<<<dim3(C3 / 32, C / 32), dim3(32, 8), 0, stream>>>(w_qkv, wqkvT, C, C3);
    transpose_convert<<<dim3(C / 32, C / 32),  dim3(32, 8), 0, stream>>>(w_o,   w_oT,  C, C);
    // 3) QKV projection (bf16 out), 128x128 block tiles
    gemm_bf16_wmma<true><<<dim3(C3 / 128, M / 128), dim3(128), 0, stream>>>(
        x_bf, wqkvT, b_qkv, qkv_bf, M, C3, C);
    // 4) skiparse flash attention: 4 subseq x 16 heads x 16 query tiles
    flash_attn<<<dim3(4 * 16 * 16), dim3(128), 0, stream>>>(qkv_bf, attn_bf);
    // 5) output projection (f32 out)
    gemm_bf16_wmma<false><<<dim3(C / 128, M / 128), dim3(128), 0, stream>>>(
        attn_bf, w_oT, b_o, out, M, C, C);
}
